// Codec_53807350284798
// MI455X (gfx1250) — matrix operand order repaired and checked by reading, not compiled, not run on hardware
//
#include <hip/hip_runtime.h>
#include <math.h>

typedef unsigned int u32;
typedef _Float16 v16h __attribute__((ext_vector_type(16)));
typedef _Float16 h2   __attribute__((ext_vector_type(2)));
typedef float    v8f  __attribute__((ext_vector_type(8)));
typedef u32      u32x8 __attribute__((ext_vector_type(8)));

#define NBLK   6144        // 8*3*16*16 blocks of 32x32
#define NPIX   6291456     // 6144*1024
#define NHID   15
#define NLAY   16          // 15 hidden + 1 output layer (Wo/bo folded in)

static __device__ __forceinline__ u32 pk2(float a, float b) {
    h2 h;
    h.x = (_Float16)a;
    h.y = (_Float16)b;
    return __builtin_bit_cast(u32, h);
}

// lane i <-> lane i^16 swap in one VALU op (v_permlanex16_b32, identity selects)
static __device__ __forceinline__ float xor16(float v) {
    int i = __builtin_bit_cast(int, v);
    int r = __builtin_amdgcn_permlanex16(i, i, 0x76543210, 0xFEDCBA98, false, false);
    return __builtin_bit_cast(float, r);
}

// exact leaky_relu(0.01): max(v, 0.01*v)  -> v_mul + v_max_num
static __device__ __forceinline__ float leaky(float v) {
    return fmaxf(v, 0.01f * v);
}

__global__ __launch_bounds__(256)
void mlp_codec_kernel(const float* __restrict__ xg,
                      const float* __restrict__ Wh,
                      const float* __restrict__ bh,
                      const float* __restrict__ Wo,
                      const float* __restrict__ bo,
                      float* __restrict__ out,
                      float* __restrict__ blockSums)
{
    __shared__ __align__(16) float lds_x[1024];          // 32x32 block, f32
    __shared__ __align__(16) u32   lds_A[NLAY * 32][8];  // per-layer A operand (WMMA A-layout)
    __shared__ __align__(16) float lds_bias[NLAY][16];   // C operand bias (layer 15 row0 = bo)
    __shared__ float wsum[8];

    const int tid = threadIdx.x;
    const int g   = blockIdx.x;
    const int bx  = g & 15;
    const int by  = (g >> 4) & 15;
    const int bc  = g >> 8;                 // b*3+c plane index, 0..23

    // ---- load 32x32 input block to LDS ----
    {
        const long plane = (long)bc * 262144;
        #pragma unroll
        for (int i = 0; i < 4; ++i) {
            int idx = tid + i * 256;
            int y = idx >> 5, xc = idx & 31;
            lds_x[idx] = xg[plane + (long)(by * 32 + y) * 512 + bx * 32 + xc];
        }
    }

    // ---- build A operands (contiguous k order k=16h+s, not AMD's table order) for 15 hidden + output layer ----
    // lane m (0-15): VGPR j -> K=2j,2j+1
    // lane m+16   : VGPR j -> K=16+2j.. (zero)
    for (int idx = tid; idx < NLAY * 32; idx += 256) {
        int l = idx >> 5, lane = idx & 31;
        int o = lane & 15, grp2 = lane >> 4;
        #pragma unroll
        for (int j = 0; j < 8; ++j) {
            int k0 = grp2 ? 99 : (2 * j);
            float va = 0.f, vb = 0.f;
            if (l < NHID) {
                if (o < 12 && k0 < 12)     va = Wh[l * 144 + o * 12 + k0];
                if (o < 12 && k0 + 1 < 12) vb = Wh[l * 144 + o * 12 + k0 + 1];
            } else {                         // output layer: only row o==0 = Wo
                if (o == 0 && k0 < 12)     va = Wo[k0];
                if (o == 0 && k0 + 1 < 12) vb = Wo[k0 + 1];
            }
            lds_A[idx][j] = pk2(va, vb);
        }
    }
    // biases folded into the WMMA C operand; layer 15 carries bo at (o==0)
    for (int idx = tid; idx < NLAY * 16; idx += 256) {
        int l = idx >> 4, o = idx & 15;
        float v;
        if (l < NHID) v = (o < 12) ? bh[l * 12 + o] : 0.f;
        else          v = (o == 0) ? bo[0] : 0.f;
        lds_bias[l][o] = v;
    }
    __syncthreads();

    const int wave = tid >> 5, lane = tid & 31;
    const int m = lane & 15;
    const u32 bmask = (lane < 16) ? 0xFFFFFFFFu : 0u;  // B rows K>=16 live in upper lanes -> zero
    const int cOff  = (lane < 16) ? 0 : 8;

    // causal neighborhood offsets (REACH=2 -> 12 features)
    const int dyt[12] = {-2,-2,-2,-2,-2,-1,-1,-1,-1,-1, 0, 0};
    const int dxt[12] = {-2,-1, 0, 1, 2,-2,-1, 0, 1, 2,-2,-1};

    float lsum = 0.f;
    const size_t outbase = 1 + (size_t)g * 1024;   // out[0] is the loss scalar

    // ---- build first-layer B operands (features) for all 8 tiles of this wave ----
    // B layout: lanes 0-15, VGPR j holds K=2j,2j+1 for pixel N=lane; lanes 16-31 zero (K>=16)
    u32 b[8][8];
    int pix[8];
    #pragma unroll
    for (int t = 0; t < 8; ++t) {
        const int p = (wave * 8 + t) * 16 + m;
        pix[t] = p;
        const int py = p >> 5, px = p & 31;
        float f[12];
        #pragma unroll
        for (int k = 0; k < 12; ++k) {
            int yy = py + dyt[k], xx = px + dxt[k];
            bool ok = (yy >= 0) && (xx >= 0) && (xx < 32);
            f[k] = ok ? lds_x[yy * 32 + xx] : 0.f;
        }
        b[t][0] = pk2(f[0],  f[1])  & bmask;
        b[t][1] = pk2(f[2],  f[3])  & bmask;
        b[t][2] = pk2(f[4],  f[5])  & bmask;
        b[t][3] = pk2(f[6],  f[7])  & bmask;
        b[t][4] = pk2(f[8],  f[9])  & bmask;
        b[t][5] = pk2(f[10], f[11]) & bmask;
        b[t][6] = 0u;
        b[t][7] = 0u;
    }

    // ---- preload layer 0 operands; software-pipeline A/C loads across layers ----
    uint4 a0, a1;
    float4 c0, c1;
    {
        const uint4* ap = (const uint4*)&lds_A[lane][0];
        a0 = ap[0]; a1 = ap[1];
        const float4* cp = (const float4*)&lds_bias[0][cOff];
        c0 = cp[0]; c1 = cp[1];
    }

    // ---- 15 hidden layers: 8 back-to-back WMMAs per layer, next layer prefetched ----
    for (int l = 0; l < NHID; ++l) {
        // prefetch layer l+1 operands (l==14 prefetches the output layer)
        const uint4* apn = (const uint4*)&lds_A[(l + 1) * 32 + lane][0];
        uint4 na0 = apn[0], na1 = apn[1];
        const float4* cpn = (const float4*)&lds_bias[l + 1][cOff];
        float4 nc0 = cpn[0], nc1 = cpn[1];

        u32x8 av = {a0.x, a0.y, a0.z, a0.w, a1.x, a1.y, a1.z, a1.w};
        v16h A = __builtin_bit_cast(v16h, av);
        v8f C = {c0.x, c0.y, c0.z, c0.w, c1.x, c1.y, c1.z, c1.w};

        #pragma unroll
        for (int t = 0; t < 8; ++t) {
            u32x8 bu = {b[t][0], b[t][1], b[t][2], b[t][3],
                        b[t][4], b[t][5], b[t][6], b[t][7]};
            v16h B = __builtin_bit_cast(v16h, bu);

            v8f D = __builtin_amdgcn_wmma_f32_16x16x32_f16(
                        false, A, false, B, (short)0, C, false, false);

            float act[8];
            #pragma unroll
            for (int j = 0; j < 8; ++j) act[j] = leaky(D[j]);

            // rows o=8..11 live in lane^16 -> one v_permlanex16 per reg
            float ax0 = xor16(act[0]), ax1 = xor16(act[1]);
            float ax2 = xor16(act[2]), ax3 = xor16(act[3]);

            b[t][0] = pk2(act[0], act[1]) & bmask;
            b[t][1] = pk2(act[2], act[3]) & bmask;
            b[t][2] = pk2(act[4], act[5]) & bmask;
            b[t][3] = pk2(act[6], act[7]) & bmask;
            b[t][4] = pk2(ax0, ax1) & bmask;
            b[t][5] = pk2(ax2, ax3) & bmask;
        }

        a0 = na0; a1 = na1; c0 = nc0; c1 = nc1;
    }

    // ---- output layer (layer 15, operands already prefetched): pred = D[0], lanes 0-15 ----
    {
        u32x8 av = {a0.x, a0.y, a0.z, a0.w, a1.x, a1.y, a1.z, a1.w};
        v16h A = __builtin_bit_cast(v16h, av);
        v8f C = {c0.x, c0.y, c0.z, c0.w, c1.x, c1.y, c1.z, c1.w};

        #pragma unroll
        for (int t = 0; t < 8; ++t) {
            u32x8 bu = {b[t][0], b[t][1], b[t][2], b[t][3],
                        b[t][4], b[t][5], b[t][6], b[t][7]};
            v16h B = __builtin_bit_cast(v16h, bu);
            v8f P = __builtin_amdgcn_wmma_f32_16x16x32_f16(
                        false, A, false, B, (short)0, C, false, false);

            float pred = fminf(1.f, fmaxf(-1.f, P[0]));
            float xv = lds_x[pix[t]];
            float d = fmodf(xv - pred + 1.0f, 2.0f) - 1.0f;
            if (lane < 16) out[outbase + pix[t]] = d;
            lsum += (lane < 16) ? d * d : 0.f;
        }
    }

    // ---- deterministic block reduction of sum of squares ----
    lsum += xor16(lsum);                       // fold upper/lower half-waves
    #pragma unroll
    for (int off = 8; off; off >>= 1) lsum += __shfl_xor(lsum, off, 32);
    if (lane == 0) wsum[wave] = lsum;
    __syncthreads();
    if (tid == 0) {
        float s = 0.f;
        #pragma unroll
        for (int w = 0; w < 8; ++w) s += wsum[w];
        blockSums[g] = s;
    }
}

__global__ __launch_bounds__(256)
void finalize_loss_kernel(const float* __restrict__ blockSums, float* __restrict__ out)
{
    __shared__ float s[256];
    float sum = 0.f;
    for (int i = threadIdx.x; i < NBLK; i += 256) sum += blockSums[i];
    s[threadIdx.x] = sum;
    __syncthreads();
    for (int k = 128; k > 0; k >>= 1) {
        if (threadIdx.x < k) s[threadIdx.x] += s[threadIdx.x + k];
        __syncthreads();
    }
    if (threadIdx.x == 0) out[0] = 255.0f * sqrtf(s[0] / (float)NPIX);
}

extern "C" void kernel_launch(void* const* d_in, const int* in_sizes, int n_in,
                              void* d_out, int out_size, void* d_ws, size_t ws_size,
                              hipStream_t stream)
{
    const float* x  = (const float*)d_in[0];
    const float* Wh = (const float*)d_in[1];
    const float* bh = (const float*)d_in[2];
    const float* Wo = (const float*)d_in[3];
    const float* bo = (const float*)d_in[4];
    float* out = (float*)d_out;
    float* blockSums = (float*)d_ws;   // 6144 floats of scratch

    mlp_codec_kernel<<<NBLK, 256, 0, stream>>>(x, Wh, bh, Wo, bo, out, blockSums);
    finalize_loss_kernel<<<1, 256, 0, stream>>>(blockSums, out);
}
